// Wasserstein_POT_9612136808913
// MI455X (gfx1250) — matrix operand order repaired and checked by reading, not compiled, not run on hardware
//
#include <hip/hip_runtime.h>
#include <hip/hip_bf16.h>

// CDNA5 (gfx1250) vector types for WMMA
typedef __attribute__((ext_vector_type(16))) __bf16 v16bf;
typedef __attribute__((ext_vector_type(8)))  __bf16 v8bf;
typedef __attribute__((ext_vector_type(8)))  float  v8f;

#define EPS_INV   20.0f      // 1 / 0.05
#define TINY_F    1e-35f
#define NCH_PAD   16         // scaling vectors padded to 16 "channels" (3 live)

// ---------------------------------------------------------------------------
// Kernel 1: build per-channel distributions a, b from the RGB inputs, and
// initialize v0 = ones (bf16 copy used as WMMA B operand).
// grid = 6 blocks: 0..2 -> channels of in0 -> a ; 3..5 -> channels of in1 -> b
// ---------------------------------------------------------------------------
__global__ __launch_bounds__(256)
void init_dists_kernel(const float* __restrict__ in0,
                       const float* __restrict__ in1,
                       float* __restrict__ a, float* __restrict__ b,
                       __bf16* __restrict__ v_bf16, float* __restrict__ v_f32,
                       int n)
{
    const int blk = blockIdx.x;          // 0..5
    const int c   = blk % 3;
    const float* x = ((blk < 3) ? in0 : in1) + (size_t)c * n;
    float* dst     = (blk < 3) ? a : b;

    __shared__ float sred[256];
    float s = 0.0f;
    for (int i = threadIdx.x; i < n; i += 256) {
        float val = fmaxf(x[i] * 0.5f + 0.5f, 0.0001f);
        s += val;
    }
    sred[threadIdx.x] = s;
    __syncthreads();
    for (int off = 128; off > 0; off >>= 1) {
        if (threadIdx.x < off) sred[threadIdx.x] += sred[threadIdx.x + off];
        __syncthreads();
    }
    const float inv = 1.0f / (sred[0] + TINY_F);

    for (int i = threadIdx.x; i < n; i += 256) {
        float val = fmaxf(x[i] * 0.5f + 0.5f, 0.0001f) * inv;
        dst[(size_t)c * n + i] = val;
        if (blk >= 3) {                       // v0 = ones (live channels only)
            v_bf16[(size_t)c * n + i] = (__bf16)1.0f;
            v_f32 [(size_t)c * n + i] = 1.0f;
        }
    }
}

// ---------------------------------------------------------------------------
// Kernel 1b: zero the padding rows (channels 3..15) of the bf16 scaling
// vectors so the WMMA B fragment can be loaded unconditionally by all lanes
// (no EXEC divergence, no VALU zero-fills -> no WMMA WAR hazard NOPs).
// Runs every launch (workspace is not re-initialized between replays).
// ---------------------------------------------------------------------------
__global__ __launch_bounds__(256)
void zero_pad_kernel(__bf16* __restrict__ u_bf16, __bf16* __restrict__ v_bf16,
                     int n)
{
    const int total = (NCH_PAD - 3) * n;           // 13*n pad elements
    const int base  = 3 * n;
    for (int i = blockIdx.x * 256 + threadIdx.x; i < total;
         i += gridDim.x * 256) {
        u_bf16[base + i] = (__bf16)0.0f;
        v_bf16[base + i] = (__bf16)0.0f;
    }
}

// ---------------------------------------------------------------------------
// Kernel 2: Gibbs kernel K = exp(-M/eps) in bf16.
// 170 MB result -> resident in the 192 MB global L2 for the Sinkhorn loop.
// ---------------------------------------------------------------------------
__global__ __launch_bounds__(256)
void build_K_kernel(const float* __restrict__ M, __bf16* __restrict__ K,
                    unsigned long long total)
{
    unsigned long long idx    = (unsigned long long)blockIdx.x * blockDim.x + threadIdx.x;
    unsigned long long stride = (unsigned long long)gridDim.x * blockDim.x;
    for (; idx < total; idx += stride)
        K[idx] = (__bf16)__expf(M[idx] * -EPS_INV);
}

// ---------------------------------------------------------------------------
// Kernel 3: one Sinkhorn half-step:  out = norm / (K @ vin + TINY)
// K symmetric, so both u- and v-updates use this kernel.
//
// Each workgroup (256 threads = 8 waves) owns 32 rows of K (two 16-row WMMA
// A tiles sharing one B fragment). Waves partition the k-dimension in
// interleaved chunks of 32 and accumulate with v_wmma_f32_16x16x32_bf16.
// Partial D tiles are reduced across waves through LDS.
//
// Fragment layouts per CDNA5 ISA 7.12.2 (wave32):
//   A 16x32 bf16 : row M = lane&15 ; lanes<16: cols {k..k+7, k+16..k+23},
//                  lanes>=16: cols {k+8..k+15, k+24..k+31}
//   B 32x16 bf16 : col N = lane&15 ; same k split as A
//   D 16x16 f32  : elem r of v8f -> M = r + 8*(lane>=16), N = lane&15
// vin is padded to 16 rows (13 zero), so every lane loads unconditionally.
// ---------------------------------------------------------------------------
__global__ __launch_bounds__(256)
void sinkhorn_matvec_kernel(const __bf16* __restrict__ K,
                            const __bf16* __restrict__ vin,
                            const float*  __restrict__ norm,
                            float*   __restrict__ out_f32,
                            __bf16*  __restrict__ out_bf16,
                            int n)
{
    const int tid  = threadIdx.x;
    const int wave = tid >> 5;
    const int lane = tid & 31;
    const int hi   = lane >> 4;          // half-wave selector
    const int col  = lane & 15;          // A: row within tile / B,D: column
    const int rowBase = blockIdx.x * 32;

    const __bf16* __restrict__ Arow0 = K + (size_t)(rowBase + col) * n;
    const __bf16* __restrict__ Arow1 = Arow0 + (size_t)16 * n;
    const __bf16* __restrict__ Bcol  = vin + (size_t)col * n;   // col 3..15 -> zeros

    v8f acc0 = {};
    v8f acc1 = {};

    // interleaved k-chunks of 32, stride 8 waves * 32 = 256; unroll x2 so the
    // compiler can pipeline two B fragments / four WMMAs per trip.
    #pragma unroll 2
    for (int k = wave * 32; k < n; k += 256) {
        // ---- B fragment: k + hi*8 .. +7 and k + 16 + hi*8 .. +7 ----
        v16bf bfrag;
        {
            const v8bf blo = *(const v8bf*)(Bcol + k + hi * 8);
            const v8bf bhi = *(const v8bf*)(Bcol + k + 16 + hi * 8);
            #pragma unroll
            for (int e = 0; e < 8; ++e) { bfrag[e] = blo[e]; bfrag[8 + e] = bhi[e]; }
        }

        // ---- A fragments: two 16-byte loads per tile ----
        v16bf afrag0, afrag1;
        {
            const v8bf c0 = *(const v8bf*)(Arow0 + k + hi * 8);
            const v8bf c1 = *(const v8bf*)(Arow0 + k + 16 + hi * 8);
            #pragma unroll
            for (int e = 0; e < 8; ++e) { afrag0[e] = c0[e]; afrag0[8 + e] = c1[e]; }
        }
        {
            const v8bf c0 = *(const v8bf*)(Arow1 + k + hi * 8);
            const v8bf c1 = *(const v8bf*)(Arow1 + k + 16 + hi * 8);
            #pragma unroll
            for (int e = 0; e < 8; ++e) { afrag1[e] = c0[e]; afrag1[8 + e] = c1[e]; }
        }

        acc0 = __builtin_amdgcn_wmma_f32_16x16x32_bf16(
                   false, afrag0, false, bfrag, (short)0, acc0, false, false);
        acc1 = __builtin_amdgcn_wmma_f32_16x16x32_bf16(
                   false, afrag1, false, bfrag, (short)0, acc1, false, false);
    }

    // ---- cross-wave reduction in LDS: [wave][tile][vgpr][lane] ----
    __shared__ float red[8 * 512];
    float* myred = red + wave * 512;
    #pragma unroll
    for (int r = 0; r < 8; ++r) {
        myred[r * 32 + lane]       = acc0[r];
        myred[256 + r * 32 + lane] = acc1[r];
    }
    __syncthreads();

    const int r = tid >> 5;       // vgpr slot 0..7
    const int l = tid & 31;       // lane slot
    #pragma unroll
    for (int t = 0; t < 2; ++t) {
        float s = 0.0f;
        #pragma unroll
        for (int w = 0; w < 8; ++w) s += red[w * 512 + t * 256 + r * 32 + l];
        const int N = l & 15;
        if (N < 3) {
            const int Mrow = rowBase + t * 16 + r + 8 * (l >> 4);
            const float val = norm[(size_t)N * n + Mrow] / (s + TINY_F);
            out_f32 [(size_t)N * n + Mrow] = val;
            out_bf16[(size_t)N * n + Mrow] = (__bf16)val;
        }
    }
}

// ---------------------------------------------------------------------------
// Kernel 4: transport cost partials. Block i owns row i; threads stride the
// columns (coalesced M reads); K*M recomputed on the fly in fp32.
// partials[i] = sum_c u[c][i] * sum_j exp(-M_ij/eps)*M_ij*v[c][j]
// ---------------------------------------------------------------------------
__global__ __launch_bounds__(256)
void cost_kernel(const float* __restrict__ M,
                 const float* __restrict__ u,
                 const float* __restrict__ v,
                 float* __restrict__ partials, int n)
{
    const int i = blockIdx.x;
    const float* __restrict__ Mi = M + (size_t)i * n;
    float s0 = 0.0f, s1 = 0.0f, s2 = 0.0f;
    for (int j = threadIdx.x; j < n; j += 256) {
        const float m = Mi[j];
        const float w = __expf(m * -EPS_INV) * m;
        s0 += w * v[j];
        s1 += w * v[(size_t)n + j];
        s2 += w * v[(size_t)2 * n + j];
    }
    float t = s0 * u[i] + s1 * u[(size_t)n + i] + s2 * u[(size_t)2 * n + i];

    __shared__ float sred[256];
    sred[threadIdx.x] = t;
    __syncthreads();
    for (int off = 128; off > 0; off >>= 1) {
        if (threadIdx.x < off) sred[threadIdx.x] += sred[threadIdx.x + off];
        __syncthreads();
    }
    if (threadIdx.x == 0) partials[i] = sred[0];
}

// ---------------------------------------------------------------------------
// Kernel 5: deterministic final reduction of row partials -> scalar output.
// ---------------------------------------------------------------------------
__global__ __launch_bounds__(256)
void finalize_kernel(const float* __restrict__ partials, float* __restrict__ out, int n)
{
    __shared__ float sred[256];
    float s = 0.0f;
    for (int i = threadIdx.x; i < n; i += 256) s += partials[i];
    sred[threadIdx.x] = s;
    __syncthreads();
    for (int off = 128; off > 0; off >>= 1) {
        if (threadIdx.x < off) sred[threadIdx.x] += sred[threadIdx.x + off];
        __syncthreads();
    }
    if (threadIdx.x == 0) out[0] = sred[0];
}

// ---------------------------------------------------------------------------
// Host-side orchestration (graph-capture safe: only async launches on stream)
// ---------------------------------------------------------------------------
extern "C" void kernel_launch(void* const* d_in, const int* in_sizes, int n_in,
                              void* d_out, int out_size, void* d_ws, size_t ws_size,
                              hipStream_t stream)
{
    (void)n_in; (void)out_size; (void)ws_size;
    const float* in0 = (const float*)d_in[0];
    const float* in1 = (const float*)d_in[1];
    const float* M   = (const float*)d_in[2];
    float* out       = (float*)d_out;

    const int n = in_sizes[0] / 3;                 // 9216 for the reference
    const unsigned long long nn = (unsigned long long)n * (unsigned long long)n;

    // ---- workspace layout ----
    char* ws = (char*)d_ws;
    size_t off = 0;
    __bf16* K       = (__bf16*)(ws + off); off += nn * sizeof(__bf16);          // ~170 MB
    float*  a_f32   = (float*) (ws + off); off += (size_t)3 * n * sizeof(float);
    float*  b_f32   = (float*) (ws + off); off += (size_t)3 * n * sizeof(float);
    float*  u_f32   = (float*) (ws + off); off += (size_t)3 * n * sizeof(float);
    float*  v_f32   = (float*) (ws + off); off += (size_t)3 * n * sizeof(float);
    __bf16* u_bf16  = (__bf16*)(ws + off); off += (size_t)NCH_PAD * n * sizeof(__bf16);
    __bf16* v_bf16  = (__bf16*)(ws + off); off += (size_t)NCH_PAD * n * sizeof(__bf16);
    float*  partial = (float*) (ws + off); off += (size_t)n * sizeof(float);

    // 1) distributions + v0 + zero padding rows of the bf16 scaling vectors
    init_dists_kernel<<<6, 256, 0, stream>>>(in0, in1, a_f32, b_f32,
                                             v_bf16, v_f32, n);
    zero_pad_kernel<<<256, 256, 0, stream>>>(u_bf16, v_bf16, n);

    // 2) K = exp(-M/eps) in bf16 (L2-resident afterwards)
    build_K_kernel<<<8192, 256, 0, stream>>>(M, K, nn);

    // 3) Sinkhorn loop: 100 x (u-update, v-update), then one more u-update
    const int mvGrid = n / 32;                     // 32 rows per workgroup
    for (int it = 0; it < 100; ++it) {
        sinkhorn_matvec_kernel<<<mvGrid, 256, 0, stream>>>(K, v_bf16, a_f32,
                                                           u_f32, u_bf16, n);
        sinkhorn_matvec_kernel<<<mvGrid, 256, 0, stream>>>(K, u_bf16, b_f32,
                                                           v_f32, v_bf16, n);
    }
    sinkhorn_matvec_kernel<<<mvGrid, 256, 0, stream>>>(K, v_bf16, a_f32,
                                                       u_f32, u_bf16, n);

    // 4) transport cost
    cost_kernel<<<n, 256, 0, stream>>>(M, u_f32, v_f32, partial, n);
    finalize_kernel<<<1, 256, 0, stream>>>(partial, out, n);
}
